// IBRNet_6468220748372
// MI455X (gfx1250) — compile-verified, matrix operand order checked by reading, not run on hardware
//
#include <hip/hip_runtime.h>
#include <hip/hip_bf16.h>

typedef __attribute__((ext_vector_type(16))) _Float16 v16h;
typedef __attribute__((ext_vector_type(8)))  _Float16 v8h;
typedef __attribute__((ext_vector_type(8)))  float    v8f;

constexpr int VIEWS    = 10;
constexpr int NF       = 35;
constexpr int NSMP     = 8;                 // samples per workgroup
constexpr int NPTS     = NSMP * VIEWS;      // 80 points = 5 WMMA row tiles
constexpr int NTHREADS = 160;               // 5 waves, one 16-row tile each

struct Params {
    const float *rgb_feat, *ray_diff, *mask, *s;
    const float *ray0W, *ray0b, *ray1W, *ray1b;
    const float *base0W, *base0b, *base1W, *base1b;
    const float *vis0W, *vis0b, *vis1W, *vis1b;
    const float *vis2aW, *vis2ab, *vis2bW, *vis2bb;
    const float *rgb0W, *rgb0b, *rgb1W, *rgb1b, *rgb2W, *rgb2b;
    float* out;
    int nsamp;
};

__device__ __forceinline__ float eluf(float x) { return x > 0.f ? x : (__expf(x) - 1.f); }
__device__ __forceinline__ float sigm(float x) { return 1.f / (1.f + __expf(-x)); }

// Load one 16-bit WMMA operand fragment for this lane.
// Layout (CDNA5 ISA 7.12.2, 16-bit A 16x32): lanes 0-15 hold K=k0..k0+7 and
// K=k0+16..k0+23 (k0 = 32*kstep); lanes 16-31 the same shifted by +8.
__device__ __forceinline__ v16h load_frag(const _Float16* row, int k0) {
    v8h a = *(const v8h*)(row + k0);
    v8h b = *(const v8h*)(row + k0 + 16);
    v16h f;
#pragma unroll
    for (int i = 0; i < 8; ++i) { f[i] = a[i]; f[i + 8] = b[i]; }
    return f;
}

// D = elu(A[16 x 32*KSTEPS] * W^T + bias), W stored [N][K] row-major in LDS.
// Each wave owns rows [rowBase, rowBase+16). In-place (Out==A) is safe: all
// A fragments are loaded into VGPRs before any store is issued.
template<int KSTEPS, int NTILES>
__device__ __forceinline__ void wmma_layer(const _Float16* A, int aStride,
                                           const _Float16* W, int wStride,
                                           const float* bias,
                                           _Float16* Out, int oStride,
                                           int rowBase, int lane) {
    const int m  = lane & 15;
    const int hi = lane >> 4;
    const int k0 = hi * 8;
    v16h afrag[KSTEPS];
#pragma unroll
    for (int ks = 0; ks < KSTEPS; ++ks)
        afrag[ks] = load_frag(A + (rowBase + m) * aStride, ks * 32 + k0);
#pragma unroll
    for (int nt = 0; nt < NTILES; ++nt) {
        v8f acc = {0.f, 0.f, 0.f, 0.f, 0.f, 0.f, 0.f, 0.f};
#pragma unroll
        for (int ks = 0; ks < KSTEPS; ++ks) {
            v16h bfrag = load_frag(W + (nt * 16 + m) * wStride, ks * 32 + k0);
            acc = __builtin_amdgcn_wmma_f32_16x16x32_f16(false, afrag[ks], false, bfrag,
                                                         (short)0, acc, false, false);
        }
        const int   col = nt * 16 + m;
        const float b   = bias[col];
#pragma unroll
        for (int r = 0; r < 8; ++r) {
            float v = eluf(acc[r] + b);
            Out[(rowBase + hi * 8 + r) * oStride + col] = (_Float16)v;
        }
    }
}

__global__ void __launch_bounds__(NTHREADS) ibrnet_kernel(Params pr) {
    // ---- f16 weight tiles, [N][Kpad] row major, zero padded -------------
    __shared__ __align__(16) _Float16 wB0[64 * 128];   // base0 (64,105)->Kpad 128
    __shared__ __align__(16) _Float16 wB1[32 * 64];    // base1 (32,64)
    __shared__ __align__(16) _Float16 wV0[32 * 32];    // vis0  (32,32)
    __shared__ __align__(16) _Float16 wV1[32 * 32];    // vis1 rows 0..31 of (33,32)
    __shared__ __align__(16) _Float16 wV2a[32 * 32];   // vis2a (32,32)
    __shared__ __align__(16) _Float16 wR0[16 * 64];    // rgb0 (16,37)->Kpad 64
    // ---- activations ----------------------------------------------------
    __shared__ __align__(16) _Float16 bufA[NPTS * 128]; // xin(105) / vis1-out / x2(64)
    __shared__ __align__(16) _Float16 bufB[NPTS * 64];  // h0(64) -> x(0..31)|tmp(32..63)
    // ---- small VALU weights ---------------------------------------------
    __shared__ _Float16 rayW0h[64];
    __shared__ _Float16 rayW1h[560];
    __shared__ float rayB0[16], rayB1[NF];
    __shared__ float sb0[64], sb1[32], sv0[32], sv1[32], sv2a[32];
    __shared__ float sv1w33[32], sv2bW[32];
    __shared__ float srgb0b[16], srgb1W[128], srgb1b[8], srgb2W[8];
    __shared__ float sc_w[NPTS], sc_mask[NPTS], sc_vis[NPTS], logit[NPTS];
    __shared__ float scal[4];  // |s|, vis1_b[32], vis2b_b, rgb2_b

    const int  tid      = threadIdx.x;
    const int  lane     = tid & 31;
    const int  rowBase  = (tid >> 5) * 16;
    const long sampBase = (long)blockIdx.x * NSMP;

    // ================= stage weights into LDS =================
    for (int i = tid; i < 64 * 128; i += NTHREADS) {
        int n = i >> 7, k = i & 127;
        wB0[i] = (_Float16)((k < 105) ? pr.base0W[n * 105 + k] : 0.f);
    }
    for (int i = tid; i < 32 * 64; i += NTHREADS) wB1[i]  = (_Float16)pr.base1W[i];
    for (int i = tid; i < 32 * 32; i += NTHREADS) wV0[i]  = (_Float16)pr.vis0W[i];
    for (int i = tid; i < 32 * 32; i += NTHREADS) wV1[i]  = (_Float16)pr.vis1W[i];
    for (int i = tid; i < 32 * 32; i += NTHREADS) wV2a[i] = (_Float16)pr.vis2aW[i];
    for (int i = tid; i < 16 * 64; i += NTHREADS) {
        int n = i >> 6, k = i & 63;
        wR0[i] = (_Float16)((k < 37) ? pr.rgb0W[n * 37 + k] : 0.f);
    }
    for (int i = tid; i < 64;  i += NTHREADS) rayW0h[i] = (_Float16)pr.ray0W[i];
    for (int i = tid; i < 560; i += NTHREADS) rayW1h[i] = (_Float16)pr.ray1W[i];
    for (int i = tid; i < 16;  i += NTHREADS) rayB0[i] = pr.ray0b[i];
    for (int i = tid; i < NF;  i += NTHREADS) rayB1[i] = pr.ray1b[i];
    for (int i = tid; i < 64;  i += NTHREADS) sb0[i] = pr.base0b[i];
    for (int i = tid; i < 32;  i += NTHREADS) {
        sb1[i] = pr.base1b[i];  sv0[i] = pr.vis0b[i];  sv1[i] = pr.vis1b[i];
        sv2a[i] = pr.vis2ab[i]; sv1w33[i] = pr.vis1W[32 * 32 + i]; sv2bW[i] = pr.vis2bW[i];
    }
    for (int i = tid; i < 16;  i += NTHREADS) srgb0b[i] = pr.rgb0b[i];
    for (int i = tid; i < 128; i += NTHREADS) srgb1W[i] = pr.rgb1W[i];
    for (int i = tid; i < 8;   i += NTHREADS) { srgb1b[i] = pr.rgb1b[i]; srgb2W[i] = pr.rgb2W[i]; }
    if (tid == 0) {
        scal[0] = fabsf(pr.s[0]); scal[1] = pr.vis1b[32];
        scal[2] = pr.vis2bb[0];   scal[3] = pr.rgb2b[0];
    }
    __syncthreads();

    // ================= phase 1: ray MLP, feat, exp_dp =================
    if (tid < NPTS) {
        const int  p = tid, ls = p / VIEWS, vv = p - ls * VIEWS;
        long gs = sampBase + ls; if (gs >= pr.nsamp) gs = pr.nsamp - 1;
        const long gp = gs * VIEWS + vv;
        float rd[4];
#pragma unroll
        for (int j = 0; j < 4; ++j) rd[j] = pr.ray_diff[gp * 4 + j];
        float h[16];
#pragma unroll
        for (int o = 0; o < 16; ++o) {
            float a = rayB0[o];
#pragma unroll
            for (int j = 0; j < 4; ++j) a += rd[j] * (float)rayW0h[o * 4 + j];
            h[o] = eluf(a);
        }
        sc_mask[p] = pr.mask[gp];
        sc_w[p]    = __expf(scal[0] * (rd[3] - 1.f));
        for (int o = 0; o < NF; ++o) {
            float a = rayB1[o];
#pragma unroll
            for (int j = 0; j < 16; ++j) a += h[j] * (float)rayW1h[o * 16 + j];
            a = eluf(a);
            bufA[p * 128 + 70 + o] = (_Float16)(pr.rgb_feat[gp * 35 + o] + a);
        }
        for (int k = 105; k < 128; ++k) bufA[p * 128 + k] = (_Float16)0.f;
    }
    __syncthreads();

    // ================= phase 2: normalize w per sample =================
    if (tid < NSMP) {
        const int pb = tid * VIEWS;
        float mn = 3.4e38f;
        for (int v = 0; v < VIEWS; ++v) mn = fminf(mn, sc_w[pb + v]);
        float sum = 0.f;
        for (int v = 0; v < VIEWS; ++v) {
            float w = (sc_w[pb + v] - mn) * sc_mask[pb + v];
            sc_w[pb + v] = w; sum += w;
        }
        const float inv = 1.f / (sum + 1e-8f);
        for (int v = 0; v < VIEWS; ++v) sc_w[pb + v] *= inv;
    }
    __syncthreads();

    // ================= phase 3: weighted mean/var per (sample, feat) ====
    for (int pair = tid; pair < NSMP * NF; pair += NTHREADS) {
        const int s2 = pair / NF, j = pair - s2 * NF, pb = s2 * VIEWS;
        float mean = 0.f;
        for (int v = 0; v < VIEWS; ++v) mean += sc_w[pb + v] * (float)bufA[(pb + v) * 128 + 70 + j];
        float var = 0.f;
        for (int v = 0; v < VIEWS; ++v) {
            float d = (float)bufA[(pb + v) * 128 + 70 + j] - mean;
            var += sc_w[pb + v] * d * d;
        }
        const _Float16 hm = (_Float16)mean, hv = (_Float16)var;
        for (int v = 0; v < VIEWS; ++v) {
            bufA[(pb + v) * 128 + j]      = hm;
            bufA[(pb + v) * 128 + 35 + j] = hv;
        }
    }
    __syncthreads();

    // ====== WMMA pipeline: each wave owns rows [rowBase, rowBase+16) ======
    // base0: 105(pad128) -> 64, elu
    wmma_layer<4, 4>(bufA, 128, wB0, 128, sb0, bufB, 64, rowBase, lane);
    // base1: 64 -> 32, elu (in place, cols 0..31 of bufB = x)
    wmma_layer<2, 2>(bufB, 64, wB1, 64, sb1, bufB, 64, rowBase, lane);

    const int prow  = rowBase + (lane & 15);
    const int chalf = (lane >> 4) * 16;

    // x * w -> tmp (bufB cols 32..63)
    {
        const float w = sc_w[prow];
#pragma unroll
        for (int c = 0; c < 16; ++c)
            bufB[prow * 64 + 32 + chalf + c] = (_Float16)((float)bufB[prow * 64 + chalf + c] * w);
    }
    // vis0: 32 -> 32, elu (in place on tmp)
    wmma_layer<1, 2>(bufB + 32, 64, wV0, 32, sv0, bufB + 32, 64, rowBase, lane);
    // vis1 (first 32 outputs = x_res) -> bufA cols 0..31, elu
    wmma_layer<1, 2>(bufB + 32, 64, wV1, 32, sv1, bufA, 128, rowBase, lane);
    // 33rd vis1 column (elu then sigmoid * mask) + residual x += x_res
    {
        float acc = 0.f;
#pragma unroll
        for (int c = 0; c < 16; ++c)
            acc += (float)bufB[prow * 64 + 32 + chalf + c] * sv1w33[chalf + c];
        acc += __shfl_xor(acc, 16);
        if (lane < 16) sc_vis[prow] = sigm(eluf(acc + scal[1])) * sc_mask[prow];
#pragma unroll
        for (int c = 0; c < 16; ++c) {
            const int col = chalf + c;
            bufB[prow * 64 + col] =
                (_Float16)((float)bufB[prow * 64 + col] + (float)bufA[prow * 128 + col]);
        }
    }
    // x * vis -> tmp
    {
        const float vs = sc_vis[prow];
#pragma unroll
        for (int c = 0; c < 16; ++c)
            bufB[prow * 64 + 32 + chalf + c] = (_Float16)((float)bufB[prow * 64 + chalf + c] * vs);
    }
    // vis2a: 32 -> 32, elu (in place on tmp)
    wmma_layer<1, 2>(bufB + 32, 64, wV2a, 32, sv2a, bufB + 32, 64, rowBase, lane);
    // vis2b: 32 -> 1, sigmoid * mask
    {
        float acc = 0.f;
#pragma unroll
        for (int c = 0; c < 16; ++c)
            acc += (float)bufB[prow * 64 + 32 + chalf + c] * sv2bW[chalf + c];
        acc += __shfl_xor(acc, 16);
        if (lane < 16) sc_vis[prow] = sigm(acc + scal[2]) * sc_mask[prow];
    }
    // assemble x2 = [x(32) | vis(1) | ray_diff(4) | 0-pad] in bufA cols 0..63
    {
#pragma unroll
        for (int c = 0; c < 16; ++c)
            bufA[prow * 128 + chalf + c] = bufB[prow * 64 + chalf + c];
        if (lane < 16) {
            bufA[prow * 128 + 32] = (_Float16)sc_vis[prow];
            const int ls = prow / VIEWS, vv = prow - ls * VIEWS;
            long gs = sampBase + ls; if (gs >= pr.nsamp) gs = pr.nsamp - 1;
            const long gp = gs * VIEWS + vv;
#pragma unroll
            for (int j = 0; j < 4; ++j)
                bufA[prow * 128 + 33 + j] = (_Float16)pr.ray_diff[gp * 4 + j];
            for (int k = 37; k < 64; ++k) bufA[prow * 128 + k] = (_Float16)0.f;
        }
    }
    // rgb0: 37(pad64) -> 16, elu -> bufB cols 32..47
    wmma_layer<2, 1>(bufA, 128, wR0, 64, srgb0b, bufB + 32, 64, rowBase, lane);

    __syncthreads();

    // ================= rgb1/rgb2 + mask -> blend logits =================
    if (tid < NPTS) {
        float g0[16];
#pragma unroll
        for (int j = 0; j < 16; ++j) g0[j] = (float)bufB[tid * 64 + 32 + j];
        float g1[8];
#pragma unroll
        for (int o = 0; o < 8; ++o) {
            float a = srgb1b[o];
#pragma unroll
            for (int j = 0; j < 16; ++j) a += g0[j] * srgb1W[o * 16 + j];
            g1[o] = eluf(a);
        }
        float a2 = scal[3];
#pragma unroll
        for (int j = 0; j < 8; ++j) a2 += g1[j] * srgb2W[j];
        logit[tid] = (sc_mask[tid] == 0.f) ? -1e9f : a2;
    }
    __syncthreads();

    // ================= softmax over views + rgb blend =================
    if (tid < NSMP && (sampBase + tid) < pr.nsamp) {
        const int pb = tid * VIEWS;
        float mx = -3.4e38f;
        for (int v = 0; v < VIEWS; ++v) mx = fmaxf(mx, logit[pb + v]);
        float e[VIEWS]; float sum = 0.f;
        for (int v = 0; v < VIEWS; ++v) { e[v] = __expf(logit[pb + v] - mx); sum += e[v]; }
        const float inv = 1.f / sum;
        const long gs = sampBase + tid;
        float o0 = 0.f, o1 = 0.f, o2 = 0.f;
        for (int v = 0; v < VIEWS; ++v) {
            const long gp = gs * VIEWS + v;
            const float w = e[v] * inv;
            o0 += w * pr.rgb_feat[gp * 35 + 0];
            o1 += w * pr.rgb_feat[gp * 35 + 1];
            o2 += w * pr.rgb_feat[gp * 35 + 2];
        }
        pr.out[gs * 3 + 0] = o0; pr.out[gs * 3 + 1] = o1; pr.out[gs * 3 + 2] = o2;
    }
}

extern "C" void kernel_launch(void* const* d_in, const int* in_sizes, int n_in,
                              void* d_out, int out_size, void* d_ws, size_t ws_size,
                              hipStream_t stream) {
    Params pr;
    pr.rgb_feat = (const float*)d_in[0];
    pr.ray_diff = (const float*)d_in[1];
    pr.mask     = (const float*)d_in[2];
    pr.s        = (const float*)d_in[3];
    pr.ray0W  = (const float*)d_in[4];  pr.ray0b  = (const float*)d_in[5];
    pr.ray1W  = (const float*)d_in[6];  pr.ray1b  = (const float*)d_in[7];
    pr.base0W = (const float*)d_in[8];  pr.base0b = (const float*)d_in[9];
    pr.base1W = (const float*)d_in[10]; pr.base1b = (const float*)d_in[11];
    pr.vis0W  = (const float*)d_in[12]; pr.vis0b  = (const float*)d_in[13];
    pr.vis1W  = (const float*)d_in[14]; pr.vis1b  = (const float*)d_in[15];
    pr.vis2aW = (const float*)d_in[16]; pr.vis2ab = (const float*)d_in[17];
    pr.vis2bW = (const float*)d_in[18]; pr.vis2bb = (const float*)d_in[19];
    // d_in[20..33] = geo/attention/layernorm/og params: dead w.r.t. the output
    pr.rgb0W  = (const float*)d_in[34]; pr.rgb0b  = (const float*)d_in[35];
    pr.rgb1W  = (const float*)d_in[36]; pr.rgb1b  = (const float*)d_in[37];
    pr.rgb2W  = (const float*)d_in[38]; pr.rgb2b  = (const float*)d_in[39];
    pr.out    = (float*)d_out;

    const int nsamp = in_sizes[2] / VIEWS;  // mask has B*S*V elements
    pr.nsamp = nsamp;

    dim3 grid((nsamp + NSMP - 1) / NSMP), block(NTHREADS);
    ibrnet_kernel<<<grid, block, 0, stream>>>(pr);
    (void)n_in; (void)out_size; (void)d_ws; (void)ws_size;
}
